// Attention_6253472383245
// MI455X (gfx1250) — matrix operand order repaired and checked by reading, not compiled, not run on hardware
//
#include <hip/hip_runtime.h>
#include <hip/hip_bf16.h>

typedef __bf16 bf16_t;
typedef __attribute__((ext_vector_type(16))) __bf16 v16bf;
typedef __attribute__((ext_vector_type(8)))  float  v8f;
typedef __attribute__((ext_vector_type(4)))  int    v4i;

#define HEADS 16
#define HDIM  64
#define BATCH 2
#define SEQ   2048
#define EMB   1024
#define NROWS (BATCH*SEQ)

__device__ __forceinline__ bf16_t f2bf(float f) { return (bf16_t)f; }

union AF { v16bf v; uint4 q[2]; };

// ---- CDNA5 async global->LDS copy (ASYNCcnt path), with safe fallback ------
#if defined(__has_builtin)
#  if __has_builtin(__builtin_amdgcn_global_load_async_to_lds_b128)
#    define HAVE_ASYNC_LDS 1
#  endif
#  if __has_builtin(__builtin_amdgcn_s_wait_asynccnt)
#    define HAVE_WAIT_ASYNC 1
#  endif
#endif

__device__ __forceinline__ void async_copy16(const bf16_t* g, bf16_t* l)
{
#ifdef HAVE_ASYNC_LDS
    __builtin_amdgcn_global_load_async_to_lds_b128(
        (__attribute__((address_space(1))) v4i*)(uintptr_t)g,
        (__attribute__((address_space(3))) v4i*)(unsigned)(uintptr_t)l,
        0, 0);
#else
    *(uint4*)l = *(const uint4*)g;
#endif
}

__device__ __forceinline__ void async_wait0()
{
#ifdef HAVE_ASYNC_LDS
#  ifdef HAVE_WAIT_ASYNC
    __builtin_amdgcn_s_wait_asynccnt(0);
#  else
    asm volatile("s_wait_asynccnt 0" ::: "memory");
#  endif
#endif
}

// ---------------------------------------------------------------------------
// Kernel 0a: x f32 -> bf16 (one shot; removes 16x redundant per-tile converts)
// ---------------------------------------------------------------------------
__global__ __launch_bounds__(256)
void cvt_x_kernel(const float* __restrict__ x, bf16_t* __restrict__ xbf)
{
    size_t i = ((size_t)blockIdx.x * 256 + threadIdx.x) * 8;
    float4 f0 = *(const float4*)(x + i);
    float4 f1 = *(const float4*)(x + i + 4);
    bf16_t o[8] = { f2bf(f0.x), f2bf(f0.y), f2bf(f0.z), f2bf(f0.w),
                    f2bf(f1.x), f2bf(f1.y), f2bf(f1.z), f2bf(f1.w) };
    *(uint4*)(xbf + i) = *(const uint4*)o;
}

// ---------------------------------------------------------------------------
// Kernel 0b: W f32 -> bf16 transposed (WT[c][k]); LDS-tiled 64x64 transpose.
// Makes B-operand staging a contiguous bf16 copy (async-LDS friendly).
// grid = (E/64, E/64, 4)
// ---------------------------------------------------------------------------
__global__ __launch_bounds__(128)
void cvt_w_kernel(const float* __restrict__ Wq, const float* __restrict__ Wk,
                  const float* __restrict__ Wv, const float* __restrict__ Wo,
                  bf16_t* __restrict__ wtq, bf16_t* __restrict__ wtk,
                  bf16_t* __restrict__ wtv, bf16_t* __restrict__ wto)
{
    __shared__ bf16_t lt[64][72];
    const int zi = blockIdx.z;
    const float* W  = (zi == 0) ? Wq : (zi == 1) ? Wk : (zi == 2) ? Wv : Wo;
    bf16_t*      WT = (zi == 0) ? wtq : (zi == 1) ? wtk : (zi == 2) ? wtv : wto;
    const int k0 = blockIdx.x * 64, c0 = blockIdx.y * 64;
    const int t = threadIdx.x;
    {
        int r = t >> 1, cs = (t & 1) * 32;              // row k, 32 cols
        const float* src = W + (size_t)(k0 + r) * EMB + c0 + cs;
#pragma unroll
        for (int i = 0; i < 32; i += 4) {
            float4 f = *(const float4*)(src + i);
            lt[cs + i + 0][r] = f2bf(f.x); lt[cs + i + 1][r] = f2bf(f.y);
            lt[cs + i + 2][r] = f2bf(f.z); lt[cs + i + 3][r] = f2bf(f.w);
        }
    }
    __syncthreads();
    {
        int c = t >> 1, ks = (t & 1) * 32;              // row c, 32 k's
        uint4* d4 = (uint4*)(WT + (size_t)(c0 + c) * EMB + k0 + ks);
        const uint4* s4 = (const uint4*)&lt[c][ks];
        d4[0] = s4[0]; d4[1] = s4[1]; d4[2] = s4[2]; d4[3] = s4[3];
    }
}

// ---------------------------------------------------------------------------
// Kernel 1: QKV projection + RoPE, double-buffered async LDS staging.
// grid=(NROWS/64, EMB/64, 3), 128 thr. Q additionally scaled by 1/sqrt(D)
// (power of two -> exact in bf16), so attention needs no score scaling.
// ---------------------------------------------------------------------------
__global__ __launch_bounds__(128)
void qkv_rope_kernel(const bf16_t* __restrict__ xbf,
                     const float* __restrict__ cosT,
                     const float* __restrict__ sinT,
                     const bf16_t* __restrict__ wtq,
                     const bf16_t* __restrict__ wtk,
                     const bf16_t* __restrict__ wtv,
                     bf16_t* __restrict__ qbuf,
                     bf16_t* __restrict__ kbuf,
                     bf16_t* __restrict__ vbuf)
{
    __shared__ bf16_t xt[2][64][48];    // A tiles: 64 rows x 32 K
    __shared__ bf16_t wt[2][64][48];    // B tiles: 64 cols x 32 K (pre-transposed)

    const int z = blockIdx.z;
    const bf16_t* WT  = (z == 0) ? wtq : ((z == 1) ? wtk : wtv);
    bf16_t*       dst = (z == 0) ? qbuf : ((z == 1) ? kbuf : vbuf);

    const int row0 = blockIdx.x * 64;
    const int col0 = blockIdx.y * 64;       // == head*64
    const int t = threadIdx.x;
    const int wave = t >> 5, lane = t & 31;
    const int lhalf = lane >> 4, l15 = lane & 15;
    const int sr = t >> 1, scs = (t & 1) * 16;   // staging coords (16 bf16/copy-pair)

    auto stage = [&](int buf, int k0) {
        const bf16_t* xs = xbf + (size_t)(row0 + sr) * EMB + k0 + scs;
        const bf16_t* ws = WT  + (size_t)(col0 + sr) * EMB + k0 + scs;
        async_copy16(xs,     &xt[buf][sr][scs]);
        async_copy16(xs + 8, &xt[buf][sr][scs + 8]);
        async_copy16(ws,     &wt[buf][sr][scs]);
        async_copy16(ws + 8, &wt[buf][sr][scs + 8]);
    };

    v8f acc[4] = {};
    stage(0, 0);
    async_wait0();
    __syncthreads();

    const int am  = wave * 16 + l15;
    const int kb8 = lhalf * 8;
    const int kb16 = lhalf * 16;

    for (int k0 = 0; k0 < EMB; k0 += 32) {
        const int cur = (k0 >> 5) & 1;
        if (k0 + 32 < EMB) stage(cur ^ 1, k0 + 32);   // async prefetch into back buffer

        AF a;
        a.q[0] = *(const uint4*)&xt[cur][am][kb16];
        a.q[1] = *(const uint4*)&xt[cur][am][kb16 + 8];
        AF b[4];
#pragma unroll
        for (int g = 0; g < 4; ++g) {
            const int bc = g * 16 + l15;
            b[g].q[0] = *(const uint4*)&wt[cur][bc][kb16];
            b[g].q[1] = *(const uint4*)&wt[cur][bc][kb16 + 8];
        }
#pragma unroll
        for (int g = 0; g < 4; ++g)
            acc[g] = __builtin_amdgcn_wmma_f32_16x16x32_bf16(
                         false, a.v, false, b[g].v, (short)0, acc[g], false, false);

        async_wait0();
        __syncthreads();
    }

    // Epilogue: RoPE (q,k) in C-fragment layout; q scaled by 1/sqrt(D)
    float outv[4][8];
#pragma unroll
    for (int g = 0; g < 4; ++g)
#pragma unroll
        for (int r = 0; r < 8; ++r) outv[g][r] = acc[g][r];

    if (z < 2) {
#pragma unroll
        for (int r = 0; r < 8; ++r) {
            int m = wave * 16 + r + lhalf * 8;
            int s = (row0 + m) & (SEQ - 1);
#pragma unroll
            for (int g = 0; g < 4; ++g) {
                int d    = g * 16 + l15;
                float c  = cosT[s * HDIM + d];
                float sn = sinT[s * HDIM + d];
                float rot = (g < 2) ? -acc[g + 2][r] : acc[g - 2][r];
                outv[g][r] = acc[g][r] * c + rot * sn;
            }
        }
    }
    if (z == 0) {
#pragma unroll
        for (int g = 0; g < 4; ++g)
#pragma unroll
            for (int r = 0; r < 8; ++r) outv[g][r] *= 0.125f;   // 1/sqrt(64)
    }

    const int h = blockIdx.y;
#pragma unroll
    for (int r = 0; r < 8; ++r) {
        int m    = wave * 16 + r + lhalf * 8;
        int rowg = row0 + m;
        int bb   = rowg >> 11;
        int s    = rowg & (SEQ - 1);
        size_t base = (((size_t)(bb * HEADS + h)) * SEQ + s) * HDIM;
#pragma unroll
        for (int g = 0; g < 4; ++g)
            dst[base + g * 16 + l15] = f2bf(outv[g][r]);
    }
}

// ---------------------------------------------------------------------------
// Kernel 2: causal flash attention, double-buffered K (async) + V (early
// register load, transpose-store after the P@V WMMAs).
// grid=(SEQ/64, HEADS, BATCH), 128 thr.
// ---------------------------------------------------------------------------
__global__ __launch_bounds__(128)
void flash_attn_kernel(const bf16_t* __restrict__ qbuf,
                       const bf16_t* __restrict__ kbuf,
                       const bf16_t* __restrict__ vbuf,
                       bf16_t* __restrict__ obuf)
{
    __shared__ bf16_t kt[2][64][72];     // K tile [kv][d]
    __shared__ bf16_t vT[2][64][72];     // V tile [d][kv] (transposed)
    __shared__ bf16_t pb[4][16][72];     // per-wave P bounce buffer

    const int qb = blockIdx.x * 64;
    const int h  = blockIdx.y;
    const int b  = blockIdx.z;
    const int t = threadIdx.x, wave = t >> 5, lane = t & 31;
    const int lhalf = lane >> 4, l15 = lane & 15;
    const size_t bh = (size_t)(b * HEADS + h) * SEQ * HDIM;
    const int sr = t >> 1, sds = (t & 1) * 32;   // staging coords

    auto stage_k = [&](int buf, int kvb) {
        const bf16_t* src = kbuf + bh + (size_t)(kvb + sr) * HDIM + sds;
        async_copy16(src,      &kt[buf][sr][sds]);
        async_copy16(src + 8,  &kt[buf][sr][sds + 8]);
        async_copy16(src + 16, &kt[buf][sr][sds + 16]);
        async_copy16(src + 24, &kt[buf][sr][sds + 24]);
    };
    auto store_v = [&](int buf, const uint4* vr) {
        bf16_t tmp[32];
        ((uint4*)tmp)[0] = vr[0]; ((uint4*)tmp)[1] = vr[1];
        ((uint4*)tmp)[2] = vr[2]; ((uint4*)tmp)[3] = vr[3];
#pragma unroll
        for (int i = 0; i < 32; ++i) vT[buf][sds + i][sr] = tmp[i];
    };

    // Preload Q A-fragments (16 rows/wave, D=64 as two K=32 chunks)
    AF aq[2];
    {
        int m = wave * 16 + l15;
        const bf16_t* qrow = qbuf + bh + (size_t)(qb + m) * HDIM;
        int kb16 = lhalf * 16;
#pragma unroll
        for (int c = 0; c < 2; ++c) {
            aq[c].q[0] = *(const uint4*)(qrow + c * 32 + kb16);
            aq[c].q[1] = *(const uint4*)(qrow + c * 32 + kb16 + 8);
        }
    }

    v8f O[4] = {};
    float mrow[8], lrow[8];
#pragma unroll
    for (int r = 0; r < 8; ++r) { mrow[r] = -__builtin_inff(); lrow[r] = 0.f; }

    // Prologue staging for tile 0
    stage_k(0, 0);
    {
        const uint4* src = (const uint4*)(vbuf + bh + (size_t)sr * HDIM + sds);
        uint4 vr[4] = { src[0], src[1], src[2], src[3] };
        store_v(0, vr);
    }
    async_wait0();
    __syncthreads();

    const int ntiles = (qb >> 6) + 1;           // causal bound
    for (int ti = 0; ti < ntiles; ++ti) {
        const int kvb = ti * 64;
        const int cur = ti & 1, nxt = cur ^ 1;
        const bool pre = (ti + 1 < ntiles);

        uint4 vr[4];
        if (pre) {
            // early V loads for next tile (latency overlapped with WMMAs below)
            const uint4* src = (const uint4*)(vbuf + bh
                                + (size_t)(kvb + 64 + sr) * HDIM + sds);
            vr[0] = src[0]; vr[1] = src[1]; vr[2] = src[2]; vr[3] = src[3];
            stage_k(nxt, kvb + 64);             // async K prefetch
        }

        // Scores: S = Q * K^T  (4 kv groups x 2 d-chunks = 8 WMMA)
        v8f sc[4] = {};
#pragma unroll
        for (int c = 0; c < 2; ++c) {
#pragma unroll
            for (int g = 0; g < 4; ++g) {
                AF bk;
                const int kv   = g * 16 + l15;
                const int kb16 = lhalf * 16;
                bk.q[0] = *(const uint4*)&kt[cur][kv][c * 32 + kb16];
                bk.q[1] = *(const uint4*)&kt[cur][kv][c * 32 + kb16 + 8];
                sc[g] = __builtin_amdgcn_wmma_f32_16x16x32_bf16(
                            false, aq[c].v, false, bk.v, (short)0, sc[g], false, false);
            }
        }

        // causal mask (scale already folded into Q)
#pragma unroll
        for (int g = 0; g < 4; ++g) {
            const int kvcol = kvb + g * 16 + l15;
#pragma unroll
            for (int r = 0; r < 8; ++r) {
                const int qrow = qb + wave * 16 + r + lhalf * 8;
                sc[g][r] = (kvcol <= qrow) ? sc[g][r] : -__builtin_inff();
            }
        }

        // online softmax in C layout (row = one VGPR idx per 16-lane half)
        float pr[4][8];
#pragma unroll
        for (int r = 0; r < 8; ++r) {
            float mx = fmaxf(fmaxf(sc[0][r], sc[1][r]), fmaxf(sc[2][r], sc[3][r]));
#pragma unroll
            for (int off = 1; off < 16; off <<= 1)
                mx = fmaxf(mx, __shfl_xor(mx, off, 32));
            float mnew  = fmaxf(mrow[r], mx);
            float alpha = __expf(mrow[r] - mnew);
            float rs = 0.f;
#pragma unroll
            for (int g = 0; g < 4; ++g) {
                float p = __expf(sc[g][r] - mnew);
                pr[g][r] = p; rs += p;
            }
#pragma unroll
            for (int off = 1; off < 16; off <<= 1)
                rs += __shfl_xor(rs, off, 32);
            lrow[r] = lrow[r] * alpha + rs;
            mrow[r] = mnew;
#pragma unroll
            for (int g = 0; g < 4; ++g) O[g][r] *= alpha;
        }

        // C-layout -> A-layout conversion of P via LDS
#pragma unroll
        for (int r = 0; r < 8; ++r)
#pragma unroll
            for (int g = 0; g < 4; ++g)
                pb[wave][r + lhalf * 8][g * 16 + l15] = f2bf(pr[g][r]);
        __syncthreads();

        // O += P @ V  (2 kv-chunks x 4 d-groups = 8 WMMA)
#pragma unroll
        for (int c = 0; c < 2; ++c) {
            AF ap;
            const int kb16 = lhalf * 16;
            ap.q[0] = *(const uint4*)&pb[wave][l15][c * 32 + kb16];
            ap.q[1] = *(const uint4*)&pb[wave][l15][c * 32 + kb16 + 8];
#pragma unroll
            for (int g = 0; g < 4; ++g) {
                AF bv;
                const int d    = g * 16 + l15;
                const int kb16 = lhalf * 16;
                bv.q[0] = *(const uint4*)&vT[cur][d][c * 32 + kb16];
                bv.q[1] = *(const uint4*)&vT[cur][d][c * 32 + kb16 + 8];
                O[g] = __builtin_amdgcn_wmma_f32_16x16x32_bf16(
                           false, ap.v, false, bv.v, (short)0, O[g], false, false);
            }
        }

        if (pre) store_v(nxt, vr);     // transpose-store next V into back buffer

        async_wait0();
        __syncthreads();
    }

    // normalize and store bf16 into flat [N, E] attn buffer (pre-Wo)
#pragma unroll
    for (int r = 0; r < 8; ++r) {
        float inv = 1.f / lrow[r];
        int m = wave * 16 + r + lhalf * 8;
        size_t rowg = (size_t)b * SEQ + qb + m;
#pragma unroll
        for (int g = 0; g < 4; ++g)
            obuf[rowg * EMB + h * HDIM + g * 16 + l15] = f2bf(O[g][r] * inv);
    }
}

// ---------------------------------------------------------------------------
// Kernel 3: output projection attn(bf16) @ WoT(bf16) -> f32, double-buffered.
// ---------------------------------------------------------------------------
__global__ __launch_bounds__(128)
void out_proj_kernel(const bf16_t* __restrict__ a,
                     const bf16_t* __restrict__ wto,
                     float* __restrict__ out)
{
    __shared__ bf16_t at[2][64][48];
    __shared__ bf16_t wt[2][64][48];

    const int row0 = blockIdx.x * 64;
    const int col0 = blockIdx.y * 64;
    const int t = threadIdx.x, wave = t >> 5, lane = t & 31;
    const int lhalf = lane >> 4, l15 = lane & 15;
    const int sr = t >> 1, scs = (t & 1) * 16;

    auto stage = [&](int buf, int k0) {
        const bf16_t* as = a   + (size_t)(row0 + sr) * EMB + k0 + scs;
        const bf16_t* ws = wto + (size_t)(col0 + sr) * EMB + k0 + scs;
        async_copy16(as,     &at[buf][sr][scs]);
        async_copy16(as + 8, &at[buf][sr][scs + 8]);
        async_copy16(ws,     &wt[buf][sr][scs]);
        async_copy16(ws + 8, &wt[buf][sr][scs + 8]);
    };

    v8f acc[4] = {};
    stage(0, 0);
    async_wait0();
    __syncthreads();

    const int am   = wave * 16 + l15;
    const int kb8  = lhalf * 8;
    const int kb16 = lhalf * 16;

    for (int k0 = 0; k0 < EMB; k0 += 32) {
        const int cur = (k0 >> 5) & 1;
        if (k0 + 32 < EMB) stage(cur ^ 1, k0 + 32);

        AF av;
        av.q[0] = *(const uint4*)&at[cur][am][kb16];
        av.q[1] = *(const uint4*)&at[cur][am][kb16 + 8];
        AF b[4];
#pragma unroll
        for (int g = 0; g < 4; ++g) {
            const int bc = g * 16 + l15;
            b[g].q[0] = *(const uint4*)&wt[cur][bc][kb16];
            b[g].q[1] = *(const uint4*)&wt[cur][bc][kb16 + 8];
        }
#pragma unroll
        for (int g = 0; g < 4; ++g)
            acc[g] = __builtin_amdgcn_wmma_f32_16x16x32_bf16(
                         false, av.v, false, b[g].v, (short)0, acc[g], false, false);

        async_wait0();
        __syncthreads();
    }

#pragma unroll
    for (int r = 0; r < 8; ++r) {
        int m = wave * 16 + r + lhalf * 8;
#pragma unroll
        for (int g = 0; g < 4; ++g)
            out[(size_t)(row0 + m) * EMB + col0 + g * 16 + l15] = acc[g][r];
    }
}

// ---------------------------------------------------------------------------
extern "C" void kernel_launch(void* const* d_in, const int* in_sizes, int n_in,
                              void* d_out, int out_size, void* d_ws, size_t ws_size,
                              hipStream_t stream)
{
    const float* x    = (const float*)d_in[0];
    const float* cosT = (const float*)d_in[1];
    const float* sinT = (const float*)d_in[2];
    const float* Wq   = (const float*)d_in[3];
    const float* Wk   = (const float*)d_in[4];
    const float* Wv   = (const float*)d_in[5];
    const float* Wo   = (const float*)d_in[6];
    float* out = (float*)d_out;

    const size_t nE   = (size_t)NROWS * EMB;                 // 4 Mi
    const size_t wE   = (size_t)EMB * EMB;                   // 1 Mi
    const size_t qkvE = (size_t)BATCH * HEADS * SEQ * HDIM;  // 4 Mi

    bf16_t* xbf  = (bf16_t*)d_ws;
    bf16_t* wtq  = xbf + nE;
    bf16_t* wtk  = wtq + wE;
    bf16_t* wtv  = wtk + wE;
    bf16_t* wto  = wtv + wE;
    bf16_t* qbuf = wto + wE;
    bf16_t* kbuf = qbuf + qkvE;
    bf16_t* vbuf = kbuf + qkvE;
    bf16_t* abuf = vbuf + qkvE;                              // [N, E] bf16

    cvt_x_kernel<<<dim3(nE / (256 * 8)), 256, 0, stream>>>(x, xbf);
    cvt_w_kernel<<<dim3(EMB / 64, EMB / 64, 4), 128, 0, stream>>>(
        Wq, Wk, Wv, Wo, wtq, wtk, wtv, wto);

    qkv_rope_kernel<<<dim3(NROWS / 64, EMB / 64, 3), 128, 0, stream>>>(
        xbf, cosT, sinT, wtq, wtk, wtv, qbuf, kbuf, vbuf);

    flash_attn_kernel<<<dim3(SEQ / 64, HEADS, BATCH), 128, 0, stream>>>(
        qbuf, kbuf, vbuf, abuf);

    out_proj_kernel<<<dim3(NROWS / 64, EMB / 64), 128, 0, stream>>>(
        abuf, wto, out);
}
